// Setformer_18657337934209
// MI455X (gfx1250) — matrix operand order repaired and checked by reading, not compiled, not run on hardware
//
#include <hip/hip_runtime.h>
#include <hip/hip_bf16.h>

// B=8192 sets, S=64, D=H=O=128. One workgroup (4 wave32) per set.
// bf16 WMMA (16x16x32) with f32 accumulation; weights pre-transposed to bf16 in d_ws.

typedef __bf16 bf16_t;
typedef __attribute__((ext_vector_type(16))) __bf16 v16bf;
typedef __attribute__((ext_vector_type(8)))  __bf16 v8bf;
typedef __attribute__((ext_vector_type(8)))  float  v8f;

// ---------------- WMMA fragment helpers (wave32, contiguous k = 16*hb + elem) ----------------

// A-matrix 16x32 bf16: lanes 0-15 -> M=lane, elems 0..15 = K 0..15
//                      lanes 16-31 -> M=lane-16, elems = K 16..31
__device__ __forceinline__ v16bf load_A_frag(const bf16_t* src, int row0, int k0,
                                             int ld, int lane) {
    const int lh = lane & 15, hb = (lane >> 4) & 1;
    const bf16_t* p = src + (row0 + lh) * ld + k0 + hb * 16;
    v8bf lo = *(const v8bf*)p;          // K offsets +0..7
    v8bf hi = *(const v8bf*)(p + 8);    // K offsets +8..15
    return __builtin_shufflevector(lo, hi, 0,1,2,3,4,5,6,7,8,9,10,11,12,13,14,15);
}

// B-matrix 32x16 bf16, loaded from TRANSPOSED storage Bt[n][k] (row stride ld):
// lanes 0-15: N=lane, VGPR j packs K=2j,2j+1 ; lanes 16-31: N=lane-16, K=16+2j,17+2j
// -> per lane a contiguous 16-element (32B) read Bt[n][k0 + hb*16 .. +15].
__device__ __forceinline__ v16bf load_B_frag(const bf16_t* srcT, int n0, int k0,
                                             int ld, int lane) {
    const int lh = lane & 15, hb = (lane >> 4) & 1;
    return *(const v16bf*)(srcT + (n0 + lh) * ld + k0 + hb * 16);
}

__device__ __forceinline__ v8f wmma_bf16(v16bf a, v16bf b, v8f c) {
    return __builtin_amdgcn_wmma_f32_16x16x32_bf16(
        /*neg_a=*/false, a, /*neg_b=*/false, b,
        /*c_mod=*/(short)0, c, /*reuse_a=*/false, /*reuse_b=*/false);
}

// One 16-row output strip: NT 16-wide N tiles, reduction depth K (multiple of 32).
template <int NT, int K>
__device__ __forceinline__ void strip_gemm(v8f* acc, const bf16_t* A, int row0, int lda,
                                           const bf16_t* Bt, int ldb, int lane) {
#pragma unroll
    for (int k0 = 0; k0 < K; k0 += 32) {
        v16bf a = load_A_frag(A, row0, k0, lda, lane);
#pragma unroll
        for (int j = 0; j < NT; ++j) {
            v16bf b = load_B_frag(Bt, j * 16, k0, ldb, lane);
            acc[j] = wmma_bf16(a, b, acc[j]);
        }
    }
}

// ---------------- Kernel 1: weights fp32 -> transposed bf16 in workspace ----------------

__global__ void __launch_bounds__(256)
prep_weights_kernel(const float* __restrict__ Wq, const float* __restrict__ Wk,
                    const float* __restrict__ Wv, const float* __restrict__ Wo,
                    bf16_t* __restrict__ wT) {
    int idx = blockIdx.x * 256 + threadIdx.x;       // 0 .. 65535
    int m = idx >> 14;                              // which matrix
    int k = (idx >> 7) & 127;                       // fan_in
    int n = idx & 127;                              // fan_out
    const float* W = (m == 0) ? Wq : (m == 1) ? Wk : (m == 2) ? Wv : Wo;
    wT[m * 16384 + n * 128 + k] = (bf16_t)W[k * 128 + n];
}

// ---------------- Kernel 2: full set-attention block, one workgroup per set ----------------

__global__ void __launch_bounds__(128)
setformer_main_kernel(const float* __restrict__ neigh, const int* __restrict__ lengths,
                      const float* __restrict__ bq, const float* __restrict__ bk,
                      const float* __restrict__ bv, const float* __restrict__ bo,
                      const bf16_t* __restrict__ wT, float* __restrict__ out) {
    // LDS map (72 KB):
    //  [0,16K)   sX   64x128 bf16  (input tile; reused for attn_out bf16)
    //  [16K,32K) sQ   64x128 bf16  (aliased per-row by sS 64x64 f32 scores)
    //  [32K,48K) sK   64x128 bf16  \ reused as sO 64x128 f32 after AV phase
    //  [48K,64K) sVt 128x 64 bf16  /
    //  [64K,72K) sAt  64x 64 bf16  (softmax probabilities)
    __shared__ __align__(32) unsigned char smem[73728];
    bf16_t* sX  = (bf16_t*)(smem);
    bf16_t* sQ  = (bf16_t*)(smem + 16384);
    float*  sS  = (float*) (smem + 16384);
    bf16_t* sK  = (bf16_t*)(smem + 32768);
    bf16_t* sVt = (bf16_t*)(smem + 49152);
    float*  sO  = (float*) (smem + 32768);
    bf16_t* sAt = (bf16_t*)(smem + 65536);

    const int b    = blockIdx.x;
    const int tid  = threadIdx.x;
    const int lane = tid & 31;
    const int wave = tid >> 5;
    const int row0 = wave * 16;           // this wave's 16-row strip
    const int lh   = lane & 15;
    const int hb   = (lane >> 4) & 1;
    const int len  = lengths[b];

    // ---- Phase 1: stage neigh tile [64x128] f32 -> bf16 LDS (coalesced float4) ----
    {
        const float4* src = (const float4*)(neigh + (size_t)b * 64 * 128);
#pragma unroll
        for (int i = 0; i < 16; ++i) {
            int v4 = tid + i * 128;            // 2048 float4 total
            float4 v = src[v4];
            int e = v4 * 4;
            sX[e + 0] = (bf16_t)v.x; sX[e + 1] = (bf16_t)v.y;
            sX[e + 2] = (bf16_t)v.z; sX[e + 3] = (bf16_t)v.w;
        }
    }
    __syncthreads();

    // ---- Phase 2: Q = X*Wq+bq, K = X*Wk+bk, V = X*Wv+bv  (bf16 strips) ----
    {
        v8f acc[8];
        // Q -> sQ row-major [s][h]
#pragma unroll
        for (int j = 0; j < 8; ++j) acc[j] = (v8f)(0.0f);
        strip_gemm<8, 128>(acc, sX, row0, 128, wT, 128, lane);
#pragma unroll
        for (int j = 0; j < 8; ++j) {
            float bb = bq[j * 16 + lh];
#pragma unroll
            for (int r = 0; r < 8; ++r)
                sQ[(row0 + r + hb * 8) * 128 + j * 16 + lh] = (bf16_t)(acc[j][r] + bb);
        }
        // K -> sK row-major [t][h]
#pragma unroll
        for (int j = 0; j < 8; ++j) acc[j] = (v8f)(0.0f);
        strip_gemm<8, 128>(acc, sX, row0, 128, wT + 16384, 128, lane);
#pragma unroll
        for (int j = 0; j < 8; ++j) {
            float bb = bk[j * 16 + lh];
#pragma unroll
            for (int r = 0; r < 8; ++r)
                sK[(row0 + r + hb * 8) * 128 + j * 16 + lh] = (bf16_t)(acc[j][r] + bb);
        }
        // V -> sVt transposed [h][t] so AV B-fragments are contiguous reads
#pragma unroll
        for (int j = 0; j < 8; ++j) acc[j] = (v8f)(0.0f);
        strip_gemm<8, 128>(acc, sX, row0, 128, wT + 32768, 128, lane);
#pragma unroll
        for (int j = 0; j < 8; ++j) {
            float bb = bv[j * 16 + lh];
#pragma unroll
            for (int r = 0; r < 8; ++r)
                sVt[(j * 16 + lh) * 64 + row0 + r + hb * 8] = (bf16_t)(acc[j][r] + bb);
        }
    }
    __syncthreads();

    // ---- Phase 3: scores = Q*K^T (B-operand = sK since scores[s,t] = sum_h q[s,h]k[t,h]) ----
    {
        v8f acc[4];
#pragma unroll
        for (int j = 0; j < 4; ++j) acc[j] = (v8f)(0.0f);
        strip_gemm<4, 128>(acc, sQ, row0, 128, sK, 128, lane);
        // write f32 scores; aliases this wave's own sQ rows (reads complete first: DS in-order)
#pragma unroll
        for (int j = 0; j < 4; ++j)
#pragma unroll
            for (int r = 0; r < 8; ++r)
                sS[(row0 + r + hb * 8) * 64 + j * 16 + lh] = acc[j][r];
    }
    __syncthreads();

    // ---- Phase 4: masked softmax over keys, one thread per query row ----
    if (tid < 64) {
        const float scale = 0.08838834764831845f;   // 1/sqrt(D=128)
        const int s = tid;
        float mx = -3.0e38f;
        for (int t = 0; t < len; ++t) mx = fmaxf(mx, sS[s * 64 + t] * scale);
        float sum = 0.0f;
        for (int t = 0; t < len; ++t) sum += __expf(sS[s * 64 + t] * scale - mx);
        float inv = 1.0f / sum;
        for (int t = 0; t < 64; ++t) {
            float p = (t < len) ? __expf(sS[s * 64 + t] * scale - mx) * inv : 0.0f;
            sAt[s * 64 + t] = (bf16_t)p;
        }
    }
    __syncthreads();

    // ---- Phase 5: attn_out = attn * V  -> bf16 into sX (X is dead) ----
    {
        v8f acc[8];
#pragma unroll
        for (int j = 0; j < 8; ++j) acc[j] = (v8f)(0.0f);
        strip_gemm<8, 64>(acc, sAt, row0, 64, sVt, 64, lane);
#pragma unroll
        for (int j = 0; j < 8; ++j)
#pragma unroll
            for (int r = 0; r < 8; ++r)
                sX[(row0 + r + hb * 8) * 128 + j * 16 + lh] = (bf16_t)acc[j][r];
    }
    __syncthreads();   // all waves done reading sVt -> safe to reuse region as sO

    // ---- Phase 6: O = attn_out * Wo + bo -> f32 sO (reuses sK/sVt space) ----
    {
        v8f acc[8];
#pragma unroll
        for (int j = 0; j < 8; ++j) acc[j] = (v8f)(0.0f);
        strip_gemm<8, 128>(acc, sX, row0, 128, wT + 49152, 128, lane);
#pragma unroll
        for (int j = 0; j < 8; ++j) {
            float bb = bo[j * 16 + lh];
#pragma unroll
            for (int r = 0; r < 8; ++r)
                sO[(row0 + r + hb * 8) * 128 + j * 16 + lh] = acc[j][r] + bb;
        }
    }
    __syncthreads();

    // ---- Phase 7: masked mean-pool over valid rows; one thread per output channel ----
    {
        float sum = 0.0f;
        for (int s = 0; s < len; ++s) sum += sO[s * 128 + tid];
        out[(size_t)b * 128 + tid] = sum / (float)len;
    }
}

// ---------------- Host launcher ----------------

extern "C" void kernel_launch(void* const* d_in, const int* in_sizes, int n_in,
                              void* d_out, int out_size, void* d_ws, size_t ws_size,
                              hipStream_t stream) {
    (void)in_sizes; (void)n_in; (void)out_size; (void)ws_size;
    const float* neigh   = (const float*)d_in[0];
    const int*   lengths = (const int*)  d_in[1];
    const float* Wq = (const float*)d_in[2]; const float* bq = (const float*)d_in[3];
    const float* Wk = (const float*)d_in[4]; const float* bk = (const float*)d_in[5];
    const float* Wv = (const float*)d_in[6]; const float* bv = (const float*)d_in[7];
    const float* Wo = (const float*)d_in[8]; const float* bo = (const float*)d_in[9];
    float* out = (float*)d_out;
    bf16_t* wT = (bf16_t*)d_ws;   // 4 * 128*128 bf16 = 128 KB

    prep_weights_kernel<<<256, 256, 0, stream>>>(Wq, Wk, Wv, Wo, wT);
    setformer_main_kernel<<<8192, 128, 0, stream>>>(neigh, lengths, bq, bk, bv, bo, wT, out);
}
